// SimplifiedMamba_62070867361926
// MI455X (gfx1250) — matrix operand order repaired and checked by reading, not compiled, not run on hardware
//
#include <hip/hip_runtime.h>
#include <cstddef>
#include <cstdint>

// ---------------------------------------------------------------------------
// Simplified Mamba on MI455X (gfx1250, wave32, WMMA).
//   GEMMs: v_wmma_f32_16x16x32_bf16, 128x128 block tile, double-buffered LDS
//   software pipeline (global loads for tile k+1 issued before the WMMAs of
//   tile k).  Scan: one lane per (b, d_inner, n_state), 8-lane xor-shuffle
//   reductions, 1 exp/lane/step.
// ---------------------------------------------------------------------------

#define BB 8
#define LL 2048
#define DM 512
#define DI 768
#define NS 8
#define MROWS (BB * LL)       // 16384

typedef __attribute__((ext_vector_type(16))) __bf16        bf16x16;
typedef __attribute__((ext_vector_type(8)))  float         floatx8;
typedef __attribute__((ext_vector_type(4)))  float         f32x4;
typedef __attribute__((ext_vector_type(4)))  unsigned int  u32x4;
typedef __attribute__((ext_vector_type(2)))  unsigned int  u32x2;

union Frag { bf16x16 v; u32x4 q[2]; };

__device__ __forceinline__ float silu_f(float v) {
    return v / (1.0f + __expf(-v));
}

// ---------------------------------------------------------------------------
// WMMA bf16 GEMM:  C(MxN) = A(MxK) * B(KxN), f32 accumulate.
//   8 waves; each wave owns 2(M) x 4(N) 16x16 tiles.  Ping-pong LDS buffers.
//   EPI==1 : SiLU epilogue, cols < DI -> x_ssm (f32), cols >= DI -> gate(bf16)
//   EPI==0 : plain f32 store to Cf (ld = N)
//   A_BF16 : A is already bf16 (y_buf), else f32 converted while staging.
// ---------------------------------------------------------------------------
template <int EPI, bool A_BF16>
__global__ __launch_bounds__(256) void gemm_wmma_kernel(
    const void* __restrict__ Ap, const float* __restrict__ Bp,
    float* __restrict__ Cf, float* __restrict__ xs_out,
    __bf16* __restrict__ gate_out, int M, int N, int K)
{
    __shared__ alignas(16) __bf16 tA[2][128][32];   // [buf][m][k]
    __shared__ alignas(16) __bf16 tB[2][128][32];   // [buf][n][k]

    const int tid  = threadIdx.x;
    const int lane = tid & 31;
    const int wv   = tid >> 5;           // 0..7
    const int wm   = (wv >> 1) * 32;     // 0,32,64,96
    const int wn   = (wv & 1) * 64;      // 0,64
    const int m0   = blockIdx.y * 128;
    const int n0   = blockIdx.x * 128;

    // staging thread mapping
    const int arow = tid >> 3;           // 0..31 (+32 per it, 4 its)
    const int acol = (tid & 7) * 4;      // 0..28
    const int bn   = (tid & 31) * 4;     // 0..124
    const int bkp  = tid >> 5;           // k-pair slot 0..7 (+8 per it, 2 its)

    const size_t aBase = (size_t)(m0 + arow) * K + acol;    // + it*32*K + kb*32
    const size_t bCol0 = (size_t)n0 + bn;                   // + k*N

    floatx8 acc[2][4];
#pragma unroll
    for (int i = 0; i < 2; ++i)
#pragma unroll
        for (int j = 0; j < 4; ++j)
            acc[i][j] = (floatx8){0.f, 0.f, 0.f, 0.f, 0.f, 0.f, 0.f, 0.f};

    // ISA 16-bit WMMA fragment layouts (wave32):
    //  A: lane m=lane&15; two 16B LDS reads at k = {off, 16+off}, off=8*(lane>=16)
    //  B: lane n=lane&15; two 16B LDS reads at k = {off, 16+off}, off=8*(lane>=16)
    const int offa = (lane & 16) ? 8 : 0;
    const int offb = (lane & 16) ? 8 : 0;
    const int mr   = lane & 15;

    // staging registers
    f32x4 Afr[4]; u32x2 Abr[4]; f32x4 B0r[2], B1r[2];

    auto load_tile = [&](int kb) {
        const size_t kOff = (size_t)kb * 32;
        if (A_BF16) {
            const __bf16* Ab = (const __bf16*)Ap;
#pragma unroll
            for (int it = 0; it < 4; ++it)
                Abr[it] = *(const u32x2*)&Ab[aBase + (size_t)it * 32 * K + kOff];
        } else {
            const float* Af = (const float*)Ap;
#pragma unroll
            for (int it = 0; it < 4; ++it)
                Afr[it] = *(const f32x4*)&Af[aBase + (size_t)it * 32 * K + kOff];
        }
#pragma unroll
        for (int it = 0; it < 2; ++it) {
            const size_t k = kOff + it * 16 + 2 * bkp;
            B0r[it] = *(const f32x4*)&Bp[k * N + bCol0];
            B1r[it] = *(const f32x4*)&Bp[(k + 1) * N + bCol0];
        }
    };

    auto store_tile = [&](int buf) {
        if (A_BF16) {
#pragma unroll
            for (int it = 0; it < 4; ++it)
                *(u32x2*)&tA[buf][arow + it * 32][acol] = Abr[it];
        } else {
#pragma unroll
            for (int it = 0; it < 4; ++it) {
                union { __bf16 h[4]; u32x2 u; } cv;
                cv.h[0] = (__bf16)Afr[it][0]; cv.h[1] = (__bf16)Afr[it][1];
                cv.h[2] = (__bf16)Afr[it][2]; cv.h[3] = (__bf16)Afr[it][3];
                *(u32x2*)&tA[buf][arow + it * 32][acol] = cv.u;
            }
        }
#pragma unroll
        for (int it = 0; it < 2; ++it) {
            const int krow = it * 16 + 2 * bkp;
#pragma unroll
            for (int j = 0; j < 4; ++j) {
                union { __bf16 h[2]; unsigned int u; } p;
                p.h[0] = (__bf16)B0r[it][j];
                p.h[1] = (__bf16)B1r[it][j];
                *(unsigned int*)&tB[buf][bn + j][krow] = p.u;
            }
        }
    };

    const int nk = K >> 5;

    // prologue: tile 0 into buffer 0
    load_tile(0);
    store_tile(0);
    __syncthreads();

    for (int kb = 0; kb < nk; ++kb) {
        const int cur = kb & 1;

        // issue next tile's global loads early (latency hidden behind WMMAs)
        if (kb + 1 < nk) {
            load_tile(kb + 1);
            if (kb + 2 < nk) {  // L2 prefetch two tiles ahead
                const size_t k2 = (size_t)(kb + 2) * 32 + 2 * bkp;
                __builtin_prefetch((const char*)&Bp[k2 * N + bCol0], 0, 1);
            }
        }

        // fragments from LDS[cur]
        Frag a[2], b[4];
#pragma unroll
        for (int mi = 0; mi < 2; ++mi) {
            int r = wm + mi * 16 + mr;
            a[mi].q[0] = *(const u32x4*)&tA[cur][r][offa];
            a[mi].q[1] = *(const u32x4*)&tA[cur][r][16 + offa];
        }
#pragma unroll
        for (int ni = 0; ni < 4; ++ni) {
            int c = wn + ni * 16 + mr;
            b[ni].q[0] = *(const u32x4*)&tB[cur][c][offb];
            b[ni].q[1] = *(const u32x4*)&tB[cur][c][16 + offb];
        }

        // 8 WMMAs per wave per K-step
#pragma unroll
        for (int mi = 0; mi < 2; ++mi)
#pragma unroll
            for (int ni = 0; ni < 4; ++ni)
                acc[mi][ni] = __builtin_amdgcn_wmma_f32_16x16x32_bf16(
                    false, a[mi].v, false, b[ni].v, (short)0, acc[mi][ni],
                    false, false);

        // convert + store next tile into the other buffer
        if (kb + 1 < nk) store_tile(cur ^ 1);
        __syncthreads();
    }

    // ---- epilogue: C layout: VGPR r -> row r + 8*(lane>=16), col lane&15 ----
    const int mb = m0 + wm + ((lane & 16) ? 8 : 0);
    const int nb = n0 + wn + mr;
#pragma unroll
    for (int mi = 0; mi < 2; ++mi) {
#pragma unroll
        for (int ni = 0; ni < 4; ++ni) {
#pragma unroll
            for (int r = 0; r < 8; ++r) {
                int row = mb + mi * 16 + r;
                int col = nb + ni * 16;
                float v = acc[mi][ni][r];
                if (EPI == 1) {
                    v = silu_f(v);
                    if (col < DI)
                        xs_out[(size_t)row * DI + col] = v;
                    else
                        gate_out[(size_t)row * DI + (col - DI)] = (__bf16)v;
                } else {
                    Cf[(size_t)row * N + col] = v;
                }
            }
        }
    }
}

// ---------------------------------------------------------------------------
// t = x_ssm @ W_x  : (16384 x 768) x (768 x 8).  One wave per output row,
// lanes stride K, full-wave xor-reduction, lane 0 writes 8 floats.
// ---------------------------------------------------------------------------
__global__ __launch_bounds__(256) void rowdot_kernel(
    const float* __restrict__ xs, const float* __restrict__ Wx,
    float* __restrict__ tbuf)
{
    const int row  = blockIdx.x * 8 + (threadIdx.x >> 5);
    const int lane = threadIdx.x & 31;
    float acc[NS];
#pragma unroll
    for (int n = 0; n < NS; ++n) acc[n] = 0.f;

    for (int k = lane; k < DI; k += 32) {
        float xv = xs[(size_t)row * DI + k];
        f32x4 w0 = *(const f32x4*)&Wx[k * NS + 0];
        f32x4 w1 = *(const f32x4*)&Wx[k * NS + 4];
        acc[0] += xv * w0[0]; acc[1] += xv * w0[1];
        acc[2] += xv * w0[2]; acc[3] += xv * w0[3];
        acc[4] += xv * w1[0]; acc[5] += xv * w1[1];
        acc[6] += xv * w1[2]; acc[7] += xv * w1[3];
    }
#pragma unroll
    for (int n = 0; n < NS; ++n) {
#pragma unroll
        for (int m = 16; m >= 1; m >>= 1)
            acc[n] += __shfl_xor(acc[n], m, 32);
    }
    if (lane == 0) {
        f32x4 o0 = {acc[0], acc[1], acc[2], acc[3]};
        f32x4 o1 = {acc[4], acc[5], acc[6], acc[7]};
        *(f32x4*)&tbuf[(size_t)row * NS + 0] = o0;
        *(f32x4*)&tbuf[(size_t)row * NS + 4] = o1;
    }
}

// ---------------------------------------------------------------------------
// Selective-scan: one lane per (b, di, n).  dt shared across the 8-lane
// state group via xor shuffles; 1 exp per lane per step; y reduced the same
// way; fused * silu(z) gate; bf16 output feeds GEMM2.
// ---------------------------------------------------------------------------
__global__ __launch_bounds__(256) void scan_kernel(
    const float* __restrict__ A_log, const float* __restrict__ D_param,
    const float* __restrict__ W_dt, const float* __restrict__ b_dt,
    const float* __restrict__ xs, const float* __restrict__ tbuf,
    const __bf16* __restrict__ gate, __bf16* __restrict__ ybuf)
{
    const int g  = blockIdx.x * blockDim.x + threadIdx.x;  // 0..B*DI*NS-1
    const int n  = g & (NS - 1);
    const int ch = g >> 3;            // 0..B*DI-1
    const int di = ch % DI;
    const int b  = ch / DI;

    const float An  = -expf(A_log[di * NS + n]);
    const float wdt = W_dt[n * DI + di];
    const float bdt = b_dt[di];
    const float Dp  = D_param[di];

    float h = 0.f;
    const size_t rbase = (size_t)b * LL;
    for (int t = 0; t < LL; ++t) {
        const size_t row = rbase + t;
        const float xv = xs[row * DI + di];
        const float tv = tbuf[row * NS + n];

        float p = tv * wdt;                 // 8-lane reduce -> dt_raw
        p += __shfl_xor(p, 1, 32);
        p += __shfl_xor(p, 2, 32);
        p += __shfl_xor(p, 4, 32);
        const float raw = p + bdt;
        const float dt  = (raw > 20.f) ? raw : log1pf(__expf(raw));  // softplus

        h = __expf(dt * An) * h + dt * xv;

        float y = h;                        // 8-lane reduce -> sum over states
        y += __shfl_xor(y, 1, 32);
        y += __shfl_xor(y, 2, 32);
        y += __shfl_xor(y, 4, 32);
        if (n == 0) {
            const float gv = (float)gate[row * DI + di];
            ybuf[row * DI + di] = (__bf16)((y + Dp * xv) * gv);
        }
    }
}

// ---------------------------------------------------------------------------
extern "C" void kernel_launch(void* const* d_in, const int* in_sizes, int n_in,
                              void* d_out, int out_size, void* d_ws,
                              size_t ws_size, hipStream_t stream)
{
    (void)in_sizes; (void)n_in; (void)out_size; (void)ws_size;
    const float* x       = (const float*)d_in[0];   // (B,L,DM)
    const float* W_in    = (const float*)d_in[1];   // (DM, 2*DI)
    const float* A_log   = (const float*)d_in[2];   // (DI, N)
    const float* D_param = (const float*)d_in[3];   // (DI,)
    const float* W_x     = (const float*)d_in[4];   // (DI, N)
    const float* W_dt    = (const float*)d_in[5];   // (N, DI)
    const float* b_dt    = (const float*)d_in[6];   // (DI,)
    const float* W_out   = (const float*)d_in[7];   // (DI, DM)
    float* out = (float*)d_out;                     // (B,L,DM) f32

    // workspace layout
    char* ws = (char*)d_ws;
    float* x_ssm = (float*)ws;                                   // M*DI f32
    size_t off = (size_t)MROWS * DI * sizeof(float);
    __bf16* gate = (__bf16*)(ws + off);                          // M*DI bf16
    off += (size_t)MROWS * DI * sizeof(__bf16);
    float* t_buf = (float*)(ws + off);                           // M*NS f32
    off += (size_t)MROWS * NS * sizeof(float);
    __bf16* y_buf = (__bf16*)(ws + off);                         // M*DI bf16

    // 1) xz = x @ W_in, SiLU both halves, split into x_ssm / gate
    {
        dim3 grid((2 * DI) / 128, MROWS / 128);
        gemm_wmma_kernel<1, false><<<grid, 256, 0, stream>>>(
            (const void*)x, W_in, nullptr, x_ssm, gate, MROWS, 2 * DI, DM);
    }
    // 2) t = x_ssm @ W_x
    rowdot_kernel<<<MROWS / 8, 256, 0, stream>>>(x_ssm, W_x, t_buf);
    // 3) selective scan + gate, bf16 out
    scan_kernel<<<(BB * DI * NS) / 256, 256, 0, stream>>>(
        A_log, D_param, W_dt, b_dt, x_ssm, t_buf, gate, y_buf);
    // 4) out = y @ W_out
    {
        dim3 grid(DM / 128, MROWS / 128);
        gemm_wmma_kernel<0, true><<<grid, 256, 0, stream>>>(
            (const void*)y_buf, W_out, out, nullptr, nullptr, MROWS, DM, DI);
    }
}
